// PackedAttentionMoT_32186484916708
// MI455X (gfx1250) — compile-verified, matrix operand order checked by reading, not run on hardware
//
#include <hip/hip_runtime.h>
#include <hip/hip_bf16.h>
#include <stdint.h>

#define H_    3584
#define ZW_   4608
#define QSZ_  3584
#define KVSZ_ 512
#define HD_   128
#define NH_   28
#define NKV_  4
#define GQ_   7
#define B_    4
#define S_    1024
#define T_    4096
#define SCALE_ 0.08838834764831843f
#define MNEG_ (-1.0e30f)

typedef __attribute__((ext_vector_type(16))) __bf16 v16bf;
typedef __attribute__((ext_vector_type(8)))  float  v8f;

union FragU { v16bf v; unsigned u[8]; uint4 q[2]; };

// -------- optional CDNA5 async global->LDS path (gated, falls back) --------
#if defined(__has_builtin)
#if __has_builtin(__builtin_amdgcn_global_load_async_to_lds_b128) && \
    __has_builtin(__builtin_amdgcn_s_wait_asynccnt)
#define HAVE_ASYNC_LDS 1
#endif
#endif
#ifndef HAVE_ASYNC_LDS
#define HAVE_ASYNC_LDS 0
#endif

#if HAVE_ASYNC_LDS
typedef int v4i_ __attribute__((vector_size(16)));
__device__ __forceinline__ void async_b128(const void* g, void* l) {
  __builtin_amdgcn_global_load_async_to_lds_b128((v4i_*)(void*)g, (v4i_*)l, 0, 0);
}
#endif

__device__ __forceinline__ unsigned short f2bu(float f) {
  union { float f; unsigned u; } c; c.f = f;
  unsigned r = c.u + 0x7fffu + ((c.u >> 16) & 1u);
  return (unsigned short)(r >> 16);
}
__device__ __forceinline__ unsigned pk2(float a, float b) {
  return (unsigned)f2bu(a) | ((unsigned)f2bu(b) << 16);
}
__device__ __forceinline__ float hmax16(float v) {
  v = fmaxf(v, __shfl_xor(v, 1, 32));
  v = fmaxf(v, __shfl_xor(v, 2, 32));
  v = fmaxf(v, __shfl_xor(v, 4, 32));
  v = fmaxf(v, __shfl_xor(v, 8, 32));
  return v;
}
__device__ __forceinline__ float hsum16(float v) {
  v += __shfl_xor(v, 1, 32);
  v += __shfl_xor(v, 2, 32);
  v += __shfl_xor(v, 4, 32);
  v += __shfl_xor(v, 8, 32);
  return v;
}

// ---------------------------------------------------------------------------
// Kernel 1: gathered QKV GEMM (per modality) + bias + RMSNorm + RoPE + scatter
// grid (4608/128=36, 2048/128=16, 2 modalities), 256 threads (8 waves)
// ---------------------------------------------------------------------------
__global__ __launch_bounds__(256) void qkv_kernel(
    const float* __restrict__ x, const float* __restrict__ cosb, const float* __restrict__ sinb,
    const int* __restrict__ tix, const int* __restrict__ vix,
    const float* __restrict__ wt, const float* __restrict__ bt,
    const float* __restrict__ wv, const float* __restrict__ bv,
    const float* __restrict__ qnt, const float* __restrict__ knt,
    const float* __restrict__ qnv, const float* __restrict__ knv,
    unsigned short* __restrict__ qbuf, unsigned short* __restrict__ kbuf,
    unsigned short* __restrict__ vbuf)
{
  __shared__ unsigned short As[128][40];   // [m][k]  row = 80B (16B aligned)
  __shared__ unsigned short Bst[128][40];  // [n][k]  transposed weights

  const int mod = blockIdx.z;
  const int* idx      = mod ? vix : tix;
  const float* w      = mod ? wv  : wt;
  const float* bias   = mod ? bv  : bt;

  const int n0 = blockIdx.x * 128;
  const int m0 = blockIdx.y * 128;
  const int tid  = threadIdx.x;
  const int wvid = tid >> 5;
  const int lane = tid & 31;
  const int h    = lane >> 4;
  const int ln   = lane & 15;
  const int h8   = h << 3;

  const int arow  = tid >> 1;
  const int ahalf = tid & 1;
  const int atok  = idx[m0 + arow];
  // B loader: pair of k-rows (2kp, 2kp+1), 8 columns each
  const int bkp = tid >> 4;          // 0..15
  const int bnb = (tid & 15) * 8;    // 0..120

  const v8f vzero = {};
  v8f acc[8];
#pragma unroll
  for (int i = 0; i < 8; ++i) acc[i] = vzero;

  for (int kt = 0; kt < 112; ++kt) {
    const int kb0 = kt * 32;
    __syncthreads();
    {
      const float4* s4 = (const float4*)(x + (size_t)atok * H_ + kb0 + ahalf * 16);
      float4 fA = s4[0], fB = s4[1], fC = s4[2], fD = s4[3];
      uint4 d0, d1;
      d0.x = pk2(fA.x, fA.y); d0.y = pk2(fA.z, fA.w);
      d0.z = pk2(fB.x, fB.y); d0.w = pk2(fB.z, fB.w);
      d1.x = pk2(fC.x, fC.y); d1.y = pk2(fC.z, fC.w);
      d1.z = pk2(fD.x, fD.y); d1.w = pk2(fD.z, fD.w);
      uint4* dst = (uint4*)&As[arow][ahalf * 16];
      dst[0] = d0;
      dst[1] = d1;
    }
    {
      const float4* r0 = (const float4*)(w + (size_t)(kb0 + 2 * bkp) * ZW_ + n0 + bnb);
      const float4* r1 = (const float4*)(w + (size_t)(kb0 + 2 * bkp + 1) * ZW_ + n0 + bnb);
      float4 a0 = r0[0], a1 = r0[1], b0 = r1[0], b1 = r1[1];
      float va[8] = {a0.x, a0.y, a0.z, a0.w, a1.x, a1.y, a1.z, a1.w};
      float vb[8] = {b0.x, b0.y, b0.z, b0.w, b1.x, b1.y, b1.z, b1.w};
#pragma unroll
      for (int j = 0; j < 8; ++j)
        *(unsigned*)&Bst[bnb + j][2 * bkp] = pk2(va[j], vb[j]);
      if (kt < 111) {
        __builtin_prefetch((const void*)(w + (size_t)(kb0 + 32 + 2 * bkp) * ZW_ + n0 + bnb), 0, 1);
        __builtin_prefetch((const void*)(x + (size_t)atok * H_ + kb0 + 32 + ahalf * 16), 0, 1);
      }
    }
    __syncthreads();

    FragU fa;
    fa.q[0] = *(const uint4*)&As[wvid * 16 + ln][h8];
    fa.q[1] = *(const uint4*)&As[wvid * 16 + ln][16 + h8];
#pragma unroll
    for (int nt = 0; nt < 8; ++nt) {
      FragU fb;
      fb.q[0] = *(const uint4*)&Bst[nt * 16 + ln][h8];
      fb.q[1] = *(const uint4*)&Bst[nt * 16 + ln][16 + h8];
      acc[nt] = __builtin_amdgcn_wmma_f32_16x16x32_bf16(
          false, fa.v, false, fb.v, (short)0, acc[nt], false, false);
    }
  }

  // ---- epilogue: this 128-wide tile covers exactly one head slot ----
  const int hs = n0 / 128;          // 0..27 q, 28..31 k, 32..35 v
  float bn[8];
#pragma unroll
  for (int nt = 0; nt < 8; ++nt) bn[nt] = bias[n0 + nt * 16 + ln];

  float gw[8];
#pragma unroll
  for (int nt = 0; nt < 8; ++nt) gw[nt] = 1.0f;
  if (hs < NH_) {
    const float* g = mod ? qnv : qnt;
#pragma unroll
    for (int nt = 0; nt < 8; ++nt) gw[nt] = g[nt * 16 + ln];
  } else if (hs < NH_ + NKV_) {
    const float* g = mod ? knv : knt;
#pragma unroll
    for (int nt = 0; nt < 8; ++nt) gw[nt] = g[nt * 16 + ln];
  }

#pragma unroll
  for (int r = 0; r < 8; ++r) {
    const int tk = idx[m0 + wvid * 16 + r + 8 * h];   // C layout: row = r + 8h
    float vals[8];
#pragma unroll
    for (int nt = 0; nt < 8; ++nt) vals[nt] = acc[nt][r] + bn[nt];

    if (hs < NH_ + NKV_) {
      float ss = 0.f;
#pragma unroll
      for (int nt = 0; nt < 8; ++nt) ss += vals[nt] * vals[nt];
      ss = hsum16(ss);
      const float inv = rsqrtf(ss * (1.0f / 128.0f) + 1e-6f);
#pragma unroll
      for (int nt = 0; nt < 8; ++nt) vals[nt] *= inv * gw[nt];
      // RoPE: column c pairs with c+64 -> acc tile nt pairs with nt+4
#pragma unroll
      for (int nt = 0; nt < 4; ++nt) {
        const int c = nt * 16 + ln;
        const float co = cosb[(size_t)tk * 64 + c];
        const float si = sinb[(size_t)tk * 64 + c];
        const float x1 = vals[nt], x2 = vals[nt + 4];
        vals[nt]     = x1 * co - x2 * si;
        vals[nt + 4] = x2 * co + x1 * si;
      }
      if (hs < NH_) {
#pragma unroll
        for (int nt = 0; nt < 8; ++nt)
          qbuf[(size_t)tk * QSZ_ + hs * HD_ + nt * 16 + ln] = f2bu(vals[nt] * SCALE_);
      } else {
#pragma unroll
        for (int nt = 0; nt < 8; ++nt)
          kbuf[(size_t)tk * KVSZ_ + (hs - NH_) * HD_ + nt * 16 + ln] = f2bu(vals[nt]);
      }
    } else {
#pragma unroll
      for (int nt = 0; nt < 8; ++nt)
        vbuf[(size_t)tk * KVSZ_ + (hs - NH_ - NKV_) * HD_ + nt * 16 + ln] = f2bu(vals[nt]);
    }
  }
}

// ---------------------------------------------------------------------------
// Kernel 2: flash attention, GQA causal. grid (8 q-tiles, 28 heads, 4 batch)
// Each wave owns 16 query rows; key tiles of 64.
// ---------------------------------------------------------------------------
__global__ __launch_bounds__(256) void attn_kernel(
    const unsigned short* __restrict__ qbuf, const unsigned short* __restrict__ kbuf,
    const unsigned short* __restrict__ vbuf, unsigned short* __restrict__ obuf)
{
  __shared__ unsigned short Ks[64][144];     // [key][d]   row = 288B (16B aligned)
  __shared__ unsigned short Vt[128][72];     // [d][key]   row = 144B (16B aligned)
  __shared__ unsigned short Pw[8][16][72];   // per-wave P [row][key]

  const int qt  = blockIdx.x;
  const int hq  = blockIdx.y;
  const int bz  = blockIdx.z;
  const int kvh = hq / GQ_;
  const int q0  = qt * 128;
  const int tid  = threadIdx.x;
  const int wvid = tid >> 5;
  const int lane = tid & 31;
  const int h    = lane >> 4;
  const int ln   = lane & 15;
  const int h8   = h << 3;

  // Q fragments held in registers: 4 k-steps covering d = 0..127
  FragU qa[4];
  {
    const size_t qbase = ((size_t)(bz * S_ + q0 + wvid * 16 + ln)) * QSZ_ + (size_t)hq * HD_;
#pragma unroll
    for (int kk = 0; kk < 4; ++kk) {
      qa[kk].q[0] = *(const uint4*)(qbuf + qbase + kk * 32 + h8);
      qa[kk].q[1] = *(const uint4*)(qbuf + qbase + kk * 32 + 16 + h8);
    }
  }

  const v8f vzero = {};
  v8f oacc[8];
#pragma unroll
  for (int i = 0; i < 8; ++i) oacc[i] = vzero;
  float mrow[8], lrow[8];
#pragma unroll
  for (int r = 0; r < 8; ++r) { mrow[r] = MNEG_; lrow[r] = 0.f; }

  const int nkt = 2 * (qt + 1);           // causal: keys up to q0+127
  const int lkr = tid >> 2;               // key row 0..63 for K loader
  const int ld0 = (tid & 3) * 32;         // d chunk for K loader
  const int vkp = tid >> 3;               // key pair 0..31 for V loader
  const int vd0 = (tid & 7) * 16;         // d chunk for V loader

  for (int kt = 0; kt < nkt; ++kt) {
    const int k0 = kt * 64;
    __syncthreads();
    {
      const unsigned short* ksrc =
          kbuf + ((size_t)(bz * S_ + k0 + lkr)) * KVSZ_ + kvh * HD_ + ld0;
#if HAVE_ASYNC_LDS
#pragma unroll
      for (int j = 0; j < 4; ++j)
        async_b128(ksrc + 8 * j, &Ks[lkr][ld0 + 8 * j]);
#else
      const uint4* ks4 = (const uint4*)ksrc;
      uint4* kdst = (uint4*)&Ks[lkr][ld0];
#pragma unroll
      for (int j = 0; j < 4; ++j) kdst[j] = ks4[j];
#endif
    }
    {
      const size_t v0 = ((size_t)(bz * S_ + k0 + 2 * vkp)) * KVSZ_ + kvh * HD_ + vd0;
      const uint4* s0 = (const uint4*)(vbuf + v0);
      const uint4* s1 = (const uint4*)(vbuf + v0 + KVSZ_);
      uint4 A0 = s0[0], A1 = s0[1], B0 = s1[0], B1 = s1[1];
      unsigned a[8] = {A0.x, A0.y, A0.z, A0.w, A1.x, A1.y, A1.z, A1.w};
      unsigned b[8] = {B0.x, B0.y, B0.z, B0.w, B1.x, B1.y, B1.z, B1.w};
#pragma unroll
      for (int j = 0; j < 8; ++j) {
        const int d = vd0 + 2 * j;
        *(unsigned*)&Vt[d][2 * vkp]     = (a[j] & 0xffffu) | (b[j] << 16);
        *(unsigned*)&Vt[d + 1][2 * vkp] = (a[j] >> 16) | (b[j] & 0xffff0000u);
      }
    }
#if HAVE_ASYNC_LDS
    __builtin_amdgcn_s_wait_asynccnt(0);
#endif
    __syncthreads();

    // scores: 16x64 per wave (4 tiles of 16x16), contract over d in 4 steps
    v8f sc[4];
#pragma unroll
    for (int st = 0; st < 4; ++st) sc[st] = vzero;
#pragma unroll
    for (int kk = 0; kk < 4; ++kk) {
#pragma unroll
      for (int st = 0; st < 4; ++st) {
        FragU kf;
        kf.q[0] = *(const uint4*)&Ks[st * 16 + ln][kk * 32 + h8];
        kf.q[1] = *(const uint4*)&Ks[st * 16 + ln][kk * 32 + 16 + h8];
        sc[st] = __builtin_amdgcn_wmma_f32_16x16x32_bf16(
            false, qa[kk].v, false, kf.v, (short)0, sc[st], false, false);
      }
    }

    // online softmax (row = r + 8h, stats replicated across the 16-lane half)
#pragma unroll
    for (int r = 0; r < 8; ++r) {
      const int qpos = q0 + wvid * 16 + r + 8 * h;
      float mv = MNEG_;
#pragma unroll
      for (int st = 0; st < 4; ++st) {
        const int kpos = k0 + st * 16 + ln;
        mv = fmaxf(mv, (kpos <= qpos) ? sc[st][r] : MNEG_);
      }
      mv = hmax16(mv);
      const float mnew  = fmaxf(mrow[r], mv);
      const float alpha = __expf(mrow[r] - mnew);
      float ps[4];
      float rs = 0.f;
#pragma unroll
      for (int st = 0; st < 4; ++st) {
        const int kpos = k0 + st * 16 + ln;
        ps[st] = (kpos <= qpos) ? __expf(sc[st][r] - mnew) : 0.f;
        rs += ps[st];
      }
      rs = hsum16(rs);
      lrow[r] = lrow[r] * alpha + rs;
      mrow[r] = mnew;
#pragma unroll
      for (int dt = 0; dt < 8; ++dt) oacc[dt][r] *= alpha;
#pragma unroll
      for (int st = 0; st < 4; ++st)
        Pw[wvid][r + 8 * h][st * 16 + ln] = f2bu(ps[st]);
    }

    // O += P @ V  (contract over 64 keys = 2 wmma k-steps)
#pragma unroll
    for (int kbi = 0; kbi < 2; ++kbi) {
      FragU pf;
      pf.q[0] = *(const uint4*)&Pw[wvid][ln][kbi * 32 + h8];
      pf.q[1] = *(const uint4*)&Pw[wvid][ln][kbi * 32 + 16 + h8];
#pragma unroll
      for (int dt = 0; dt < 8; ++dt) {
        FragU vf;
        vf.q[0] = *(const uint4*)&Vt[dt * 16 + ln][kbi * 32 + h8];
        vf.q[1] = *(const uint4*)&Vt[dt * 16 + ln][kbi * 32 + 16 + h8];
        oacc[dt] = __builtin_amdgcn_wmma_f32_16x16x32_bf16(
            false, pf.v, false, vf.v, (short)0, oacc[dt], false, false);
      }
    }
  }

#pragma unroll
  for (int r = 0; r < 8; ++r) {
    const float inv = 1.0f / lrow[r];
    const size_t base =
        ((size_t)(bz * S_ + q0 + wvid * 16 + r + 8 * h)) * QSZ_ + (size_t)hq * HD_;
#pragma unroll
    for (int dt = 0; dt < 8; ++dt)
      obuf[base + dt * 16 + ln] = f2bu(oacc[dt][r] * inv);
  }
}

// ---------------------------------------------------------------------------
// Kernel 3: gathered output projection (per modality), scatter f32 to d_out
// grid (3584/128=28, 2048/128=16, 2), 256 threads
// ---------------------------------------------------------------------------
__global__ __launch_bounds__(256) void proj_kernel(
    const unsigned short* __restrict__ obuf,
    const int* __restrict__ tix, const int* __restrict__ vix,
    const float* __restrict__ wot, const float* __restrict__ wov,
    float* __restrict__ out)
{
  __shared__ unsigned short As[128][40];
  __shared__ unsigned short Bst[128][40];

  const int mod = blockIdx.z;
  const int* idx  = mod ? vix : tix;
  const float* w  = mod ? wov : wot;

  const int n0 = blockIdx.x * 128;
  const int m0 = blockIdx.y * 128;
  const int tid  = threadIdx.x;
  const int wvid = tid >> 5;
  const int lane = tid & 31;
  const int h    = lane >> 4;
  const int ln   = lane & 15;
  const int h8   = h << 3;

  const int arow  = tid >> 1;
  const int ahalf = tid & 1;
  const int atok  = idx[m0 + arow];
  const int bkp = tid >> 4;
  const int bnb = (tid & 15) * 8;

  const v8f vzero = {};
  v8f acc[8];
#pragma unroll
  for (int i = 0; i < 8; ++i) acc[i] = vzero;

  for (int kt = 0; kt < 112; ++kt) {
    const int kb0 = kt * 32;
    __syncthreads();
    {
      const unsigned short* src = obuf + (size_t)atok * QSZ_ + kb0 + ahalf * 16;
#if HAVE_ASYNC_LDS
      async_b128(src,     &As[arow][ahalf * 16]);
      async_b128(src + 8, &As[arow][ahalf * 16 + 8]);
#else
      const uint4* s4 = (const uint4*)src;
      uint4* dst = (uint4*)&As[arow][ahalf * 16];
      dst[0] = s4[0];
      dst[1] = s4[1];
#endif
    }
    {
      const float4* r0 = (const float4*)(w + (size_t)(kb0 + 2 * bkp) * H_ + n0 + bnb);
      const float4* r1 = (const float4*)(w + (size_t)(kb0 + 2 * bkp + 1) * H_ + n0 + bnb);
      float4 a0 = r0[0], a1 = r0[1], b0 = r1[0], b1 = r1[1];
      float va[8] = {a0.x, a0.y, a0.z, a0.w, a1.x, a1.y, a1.z, a1.w};
      float vb[8] = {b0.x, b0.y, b0.z, b0.w, b1.x, b1.y, b1.z, b1.w};
#pragma unroll
      for (int j = 0; j < 8; ++j)
        *(unsigned*)&Bst[bnb + j][2 * bkp] = pk2(va[j], vb[j]);
      if (kt < 111) {
        __builtin_prefetch((const void*)(w + (size_t)(kb0 + 32 + 2 * bkp) * H_ + n0 + bnb), 0, 1);
        __builtin_prefetch((const void*)(obuf + (size_t)atok * QSZ_ + kb0 + 32 + ahalf * 16), 0, 1);
      }
    }
#if HAVE_ASYNC_LDS
    __builtin_amdgcn_s_wait_asynccnt(0);
#endif
    __syncthreads();

    FragU fa;
    fa.q[0] = *(const uint4*)&As[wvid * 16 + ln][h8];
    fa.q[1] = *(const uint4*)&As[wvid * 16 + ln][16 + h8];
#pragma unroll
    for (int nt = 0; nt < 8; ++nt) {
      FragU fb;
      fb.q[0] = *(const uint4*)&Bst[nt * 16 + ln][h8];
      fb.q[1] = *(const uint4*)&Bst[nt * 16 + ln][16 + h8];
      acc[nt] = __builtin_amdgcn_wmma_f32_16x16x32_bf16(
          false, fa.v, false, fb.v, (short)0, acc[nt], false, false);
    }
  }

#pragma unroll
  for (int r = 0; r < 8; ++r) {
    const int tk = idx[m0 + wvid * 16 + r + 8 * h];
#pragma unroll
    for (int nt = 0; nt < 8; ++nt)
      out[(size_t)tk * H_ + n0 + nt * 16 + ln] = acc[nt][r];
  }
}

// ---------------------------------------------------------------------------
extern "C" void kernel_launch(void* const* d_in, const int* in_sizes, int n_in,
                              void* d_out, int out_size, void* d_ws, size_t ws_size,
                              hipStream_t stream) {
  (void)in_sizes; (void)n_in; (void)out_size; (void)ws_size;

  const float* x     = (const float*)d_in[0];
  const float* cosb  = (const float*)d_in[1];
  const float* sinb  = (const float*)d_in[2];
  /* d_in[3] = query_lens (always S, unused) */
  const int*   tix   = (const int*)d_in[4];
  const int*   vix   = (const int*)d_in[5];
  const float* wqkvt = (const float*)d_in[6];
  const float* bqkvt = (const float*)d_in[7];
  const float* wqkvv = (const float*)d_in[8];
  const float* bqkvv = (const float*)d_in[9];
  const float* wot   = (const float*)d_in[10];
  const float* wov   = (const float*)d_in[11];
  const float* qnt   = (const float*)d_in[12];
  const float* knt   = (const float*)d_in[13];
  const float* qnv   = (const float*)d_in[14];
  const float* knv   = (const float*)d_in[15];
  float* out = (float*)d_out;

  // workspace layout (bf16 stored as u16): q | k | v | attn-out  (~67 MB)
  unsigned short* qbuf = (unsigned short*)d_ws;
  unsigned short* kbuf = qbuf + (size_t)T_ * QSZ_;
  unsigned short* vbuf = kbuf + (size_t)T_ * KVSZ_;
  unsigned short* obuf = vbuf + (size_t)T_ * KVSZ_;

  qkv_kernel<<<dim3(36, 16, 2), 256, 0, stream>>>(
      x, cosb, sinb, tix, vix, wqkvt, bqkvt, wqkvv, bqkvv,
      qnt, knt, qnv, knv, qbuf, kbuf, vbuf);
  attn_kernel<<<dim3(8, NH_, B_), 256, 0, stream>>>(qbuf, kbuf, vbuf, obuf);
  proj_kernel<<<dim3(28, 16, 2), 256, 0, stream>>>(obuf, tix, vix, wot, wov, out);
}
